// UnifiedModalEncoder_44684839748099
// MI455X (gfx1250) — matrix operand order repaired and checked by reading, not compiled, not run on hardware
//
#include <hip/hip_runtime.h>
#include <hip/hip_bf16.h>
#include <math.h>

typedef __bf16 bf16;
typedef __attribute__((ext_vector_type(16))) __bf16 bf16x16;
typedef __attribute__((ext_vector_type(8)))  __bf16 bf16x8;
typedef __attribute__((ext_vector_type(8)))  float  f32x8;

// ---------------------------------------------------------------------------
// Generic batched WMMA GEMM:  C = epilogue( A[M,K](bf16) * B[K,N] + bias )
// B is supplied TRANSPOSED: BT[N,K] row-major, so both A and BT fragment loads
// are b128-pair loads (K 8h..8h+7 and 16+8h..16+8h+7, h = lane/16) matching the CDNA5 bf16 WMMA
// VGPR layout (ISA 7.12.2). Each wave computes a 32 x (16*NJ) tile
// (2 x NJ of 16x16x32 WMMAs). Block = 8 waves (2 in N, 4 in M) covering
// 128(M) x 32*NJ(N). Two-level batching: z = z1*batch2 + z2.
// ---------------------------------------------------------------------------
template<int NJ, bool GELU, bool ACC, bool STORE_BF16, bool STORE_F32>
__global__ void __launch_bounds__(256)
k_gemm(const bf16* __restrict__ A, const bf16* __restrict__ BT,
       const float* __restrict__ bias,
       float* __restrict__ Cf, bf16* __restrict__ Cb,
       int M, int N, int K, int lda, int ldb, int ldc, float alpha,
       const float* __restrict__ rowScale, int rsStride,
       int batch2,
       long long sA1, long long sA2, long long sB1, long long sB2,
       long long sC1, long long sC2)
{
    const int z  = blockIdx.z;
    const int z1 = z / batch2, z2 = z % batch2;
    A  += (long long)z1 * sA1 + (long long)z2 * sA2;
    BT += (long long)z1 * sB1 + (long long)z2 * sB2;
    const long long cOff = (long long)z1 * sC1 + (long long)z2 * sC2;

    const int wid  = threadIdx.x >> 5;            // 0..7
    const int lane = threadIdx.x & 31;
    const int wx = wid & 1, wy = wid >> 1;        // 2 x 4 wave grid
    const int mBase = blockIdx.y * 128 + wy * 32;
    const int nBase = blockIdx.x * (32 * NJ) + wx * (16 * NJ);
    const int r  = lane & 15;
    const int hi = lane >> 4;

    const f32x8 zero = {0.f,0.f,0.f,0.f,0.f,0.f,0.f,0.f};
    f32x8 acc[2][NJ];
#pragma unroll
    for (int i = 0; i < 2; ++i)
#pragma unroll
        for (int j = 0; j < NJ; ++j) acc[i][j] = zero;

    for (int k0 = 0; k0 < K; k0 += 32) {
        bf16x16 a[2], b[NJ];
#pragma unroll
        for (int i = 0; i < 2; ++i) {
            // A fragment: lane<16 holds K[k0..k0+7] & K[k0+16..k0+23],
            //             lane>=16 holds K[k0+8..15] & K[k0+24..31]
            const bf16* ap = A + (long long)(mBase + i*16 + r) * lda + k0;
            union { bf16x16 v; bf16x8 h[2]; } ua;
            ua.h[0] = *(const bf16x8*)(ap + hi * 8);
            ua.h[1] = *(const bf16x8*)(ap + 16 + hi * 8);
            a[i] = ua.v;
        }
#pragma unroll
        for (int j = 0; j < NJ; ++j) {
            // B fragment from BT[N][K]: lane<16 -> K[k0..+7] & K[k0+16..+23], lane>=16 -> K[k0+8..+15] & K[k0+24..+31]
            const bf16* bp = BT + (long long)(nBase + j*16 + r) * ldb + k0;
            union { bf16x16 v; bf16x8 h[2]; } ub;
            ub.h[0] = *(const bf16x8*)(bp + hi * 8);
            ub.h[1] = *(const bf16x8*)(bp + 16 + hi * 8);
            b[j] = ub.v;
        }
        if (k0 + 32 < K) {  // gfx1250 global_prefetch_b8 for next K slab
            __builtin_prefetch(A  + (long long)(mBase + r) * lda + k0 + 32, 0, 1);
            __builtin_prefetch(BT + (long long)(nBase + r) * ldb + k0 + 32, 0, 1);
        }
#pragma unroll
        for (int i = 0; i < 2; ++i)
#pragma unroll
            for (int j = 0; j < NJ; ++j)
                acc[i][j] = __builtin_amdgcn_wmma_f32_16x16x32_bf16(
                    false, a[i], false, b[j], (short)0, acc[i][j], false, false);
    }

    // ---- epilogue (hoisted bias / rowScale loads) ----
    // C/D layout: VGPR v, lane -> (M = v + 8*(lane>>4), N = lane&15)
    float bcol[NJ];
#pragma unroll
    for (int j = 0; j < NJ; ++j)
        bcol[j] = bias ? bias[nBase + j * 16 + r] : 0.0f;

    float rsc[2][8];
#pragma unroll
    for (int i = 0; i < 2; ++i)
#pragma unroll
        for (int v = 0; v < 8; ++v)
            rsc[i][v] = rowScale
                ? rowScale[(long long)(mBase + i * 16 + hi * 8 + v) * rsStride]
                : 1.0f;

#pragma unroll
    for (int i = 0; i < 2; ++i)
#pragma unroll
        for (int j = 0; j < NJ; ++j) {
            const int n = nBase + j * 16 + r;
#pragma unroll
            for (int v = 0; v < 8; ++v) {
                const int m = mBase + i * 16 + hi * 8 + v;
                float t = alpha * acc[i][j][v] + bcol[j];
                if (GELU) t = 0.5f * t * (1.0f + erff(t * 0.70710678118654752f));
                t *= rsc[i][v];
                const long long idx = cOff + (long long)m * ldc + n;
                if (STORE_F32) { if (ACC) Cf[idx] += t; else Cf[idx] = t; }
                if (STORE_BF16) Cb[idx] = (bf16)t;
            }
        }
}

// f32 [K][N] -> bf16 [N][K]  (weight transpose + downconvert)
__global__ void k_wT(const float* __restrict__ in, bf16* __restrict__ out,
                     int K, int N)
{
    const long long total = (long long)K * N;
    for (long long i = blockIdx.x * 256LL + threadIdx.x; i < total;
         i += (long long)gridDim.x * 256) {
        long long n = i / K, k = i % K;
        out[i] = (bf16)in[k * (long long)N + n];
    }
}

// LayerNorm over D; optional bf16 and/or f32 outputs. One block per row.
__global__ void __launch_bounds__(256)
k_ln(const float* __restrict__ x, const float* __restrict__ g,
     const float* __restrict__ b, bf16* __restrict__ outb,
     float* __restrict__ outf, int D)
{
    const long long row = blockIdx.x;
    const float* xr = x + row * D;
    __shared__ float red[256];
    float s = 0.f;
    for (int i = threadIdx.x; i < D; i += 256) s += xr[i];
    red[threadIdx.x] = s; __syncthreads();
    for (int o = 128; o > 0; o >>= 1) { if (threadIdx.x < o) red[threadIdx.x] += red[threadIdx.x + o]; __syncthreads(); }
    const float mean = red[0] / D; __syncthreads();
    float vs = 0.f;
    for (int i = threadIdx.x; i < D; i += 256) { float d = xr[i] - mean; vs += d * d; }
    red[threadIdx.x] = vs; __syncthreads();
    for (int o = 128; o > 0; o >>= 1) { if (threadIdx.x < o) red[threadIdx.x] += red[threadIdx.x + o]; __syncthreads(); }
    const float rstd = rsqrtf(red[0] / D + 1e-5f);
    for (int i = threadIdx.x; i < D; i += 256) {
        float t = (xr[i] - mean) * rstd * g[i] + b[i];
        if (outb) outb[row * D + i] = (bf16)t;
        if (outf) outf[row * D + i] = t;
    }
}

// Softmax of attention rows (bf16 in place) + mean over heads -> f32.
// One block per (b, qrow); S == 1024, 4 columns per thread.
__global__ void __launch_bounds__(256)
k_softmax_att(bf16* __restrict__ att, float* __restrict__ attMean, int S, int Hh)
{
    const int b = blockIdx.x / S, q = blockIdx.x % S;
    __shared__ float red[256];
    float macc[4] = {0.f, 0.f, 0.f, 0.f};
    for (int h = 0; h < Hh; ++h) {
        bf16* rowp = att + ((((long long)b * Hh + h) * S) + q) * (long long)S;
        float vloc[4]; float mx = -3.4e38f;
#pragma unroll
        for (int t = 0; t < 4; ++t) { vloc[t] = (float)rowp[threadIdx.x + t * 256]; mx = fmaxf(mx, vloc[t]); }
        red[threadIdx.x] = mx; __syncthreads();
        for (int o = 128; o > 0; o >>= 1) { if (threadIdx.x < o) red[threadIdx.x] = fmaxf(red[threadIdx.x], red[threadIdx.x + o]); __syncthreads(); }
        mx = red[0]; __syncthreads();
        float sum = 0.f;
#pragma unroll
        for (int t = 0; t < 4; ++t) { vloc[t] = __expf(vloc[t] - mx); sum += vloc[t]; }
        red[threadIdx.x] = sum; __syncthreads();
        for (int o = 128; o > 0; o >>= 1) { if (threadIdx.x < o) red[threadIdx.x] += red[threadIdx.x + o]; __syncthreads(); }
        const float inv = 1.0f / red[0]; __syncthreads();
#pragma unroll
        for (int t = 0; t < 4; ++t) { float p = vloc[t] * inv; rowp[threadIdx.x + t * 256] = (bf16)p; macc[t] += p; }
    }
    float* mp = attMean + ((long long)b * S + q) * (long long)S;
#pragma unroll
    for (int t = 0; t < 4; ++t) mp[threadIdx.x + t * 256] = macc[t] / Hh;
}

// v[b,s,h*64+d] (bf16) -> vT[b,h,d,s] (bf16)
__global__ void k_vtrans(const bf16* __restrict__ v, bf16* __restrict__ vT,
                         int B, int S, int Hh, int HD)
{
    const long long total = (long long)B * Hh * HD * S;
    const long long D = (long long)Hh * HD;
    for (long long i = blockIdx.x * 256LL + threadIdx.x; i < total;
         i += (long long)gridDim.x * 256) {
        long long s = i % S, d = (i / S) % HD, h = (i / (S * HD)) % Hh, b = i / (S * HD * Hh);
        vT[i] = v[(b * S + s) * D + h * HD + d];
    }
}

__global__ void k_copy_f32(const float* __restrict__ in, float* __restrict__ out, long long n)
{
    for (long long i = blockIdx.x * 256LL + threadIdx.x; i < n; i += (long long)gridDim.x * 256)
        out[i] = in[i];
}

// Router logits: one thread per (row, e)
__global__ void k_router(const float* __restrict__ xn, const float* __restrict__ Wr,
                         const float* __restrict__ br, float* __restrict__ logits,
                         int D, int E, long long rows)
{
    const long long idx = blockIdx.x * 256LL + threadIdx.x;
    if (idx >= rows * E) return;
    const long long row = idx / E; const int e = (int)(idx % E);
    float s = br[e];
    const float* xr = xn + row * D;
    for (int d = 0; d < D; ++d) s += xr[d] * Wr[(long long)d * E + e];
    logits[row * E + e] = s;
}

// softmax over E, write probs; top-2 renormalized weights scattered into comb
__global__ void k_topk(const float* __restrict__ logits, float* __restrict__ probs,
                       float* __restrict__ comb, int E, long long rows)
{
    const long long row = blockIdx.x * 256LL + threadIdx.x;
    if (row >= rows) return;
    float l[8], p[8];
    float mx = -3.4e38f;
    for (int e = 0; e < E; ++e) { l[e] = logits[row * E + e]; mx = fmaxf(mx, l[e]); }
    float sum = 0.f;
    for (int e = 0; e < E; ++e) { p[e] = __expf(l[e] - mx); sum += p[e]; }
    const float inv = 1.0f / sum;
    for (int e = 0; e < E; ++e) { p[e] *= inv; probs[row * E + e] = p[e]; }
    int i0 = 0; for (int e = 1; e < E; ++e) if (p[e] > p[i0]) i0 = e;
    int i1 = (i0 == 0) ? 1 : 0;
    for (int e = 0; e < E; ++e) if (e != i0 && p[e] > p[i1]) i1 = e;
    const float a = p[i0], bb = p[i1];     // softmax over the two top probs
    const float eb = __expf(bb - a);
    const float w0 = 1.0f / (1.0f + eb), w1 = eb / (1.0f + eb);
    float c[8]; for (int e = 0; e < E; ++e) c[e] = 0.f;
    c[i0] = w0; c[i1] = w1;
    for (int e = 0; e < E; ++e) comb[row * E + e] = c[e];
}

__global__ void k_modmask(const int* __restrict__ mt, float* __restrict__ vm,
                          float* __restrict__ tm, long long n)
{
    const long long i = blockIdx.x * 256LL + threadIdx.x;
    if (i >= n) return;
    vm[i] = (mt[i] == 0) ? 1.0f : 0.0f;
    tm[i] = (mt[i] == 1) ? 1.0f : 0.0f;
}

// ---------------------------------------------------------------------------
extern "C" void kernel_launch(void* const* d_in, const int* in_sizes, int n_in,
                              void* d_out, int out_size, void* d_ws, size_t ws_size,
                              hipStream_t stream)
{
    (void)in_sizes; (void)n_in; (void)out_size; (void)ws_size;
    const int B_ = 4, S_ = 1024, D_ = 1024, F_ = 4096, E_ = 8, H_ = 16, HD_ = 64;
    const long long M_ = (long long)B_ * S_;            // 4096 tokens

    // ---- inputs ----
    const float* x    = (const float*)d_in[0];
    const int*   mt   = (const int*)d_in[1];
    const float* n1g  = (const float*)d_in[2];  const float* n1b = (const float*)d_in[3];
    const float* n2g  = (const float*)d_in[4];  const float* n2b = (const float*)d_in[5];
    const float* Wq   = (const float*)d_in[6];  const float* bq  = (const float*)d_in[7];
    const float* Wk   = (const float*)d_in[8];  const float* bk  = (const float*)d_in[9];
    const float* Wv   = (const float*)d_in[10]; const float* bv  = (const float*)d_in[11];
    const float* Wo   = (const float*)d_in[12]; const float* bo  = (const float*)d_in[13];
    const float* Wr   = (const float*)d_in[14]; const float* br  = (const float*)d_in[15];
    const float* gen_lng = (const float*)d_in[16]; const float* gen_lnb = (const float*)d_in[17];
    const float* gen_W1  = (const float*)d_in[18]; const float* gen_b1  = (const float*)d_in[19];
    const float* gen_W2  = (const float*)d_in[20]; const float* gen_b2  = (const float*)d_in[21];

    // ---- workspace layout ----
    size_t off = 0;
    auto alloc = [&](size_t bytes) -> void* {
        void* p = (char*)d_ws + off;
        off = (off + bytes + 255) & ~(size_t)255;
        return p;
    };
    bf16* WqT = (bf16*)alloc((size_t)D_ * D_ * 2);
    bf16* WkT = (bf16*)alloc((size_t)D_ * D_ * 2);
    bf16* WvT = (bf16*)alloc((size_t)D_ * D_ * 2);
    bf16* WoT = (bf16*)alloc((size_t)D_ * D_ * 2);
    bf16* expW1T[3]; bf16* expW2T[3];
    for (int p = 0; p < 3; ++p) {
        expW1T[p] = (bf16*)alloc((size_t)D_ * F_ * 2);   // [F][D]
        expW2T[p] = (bf16*)alloc((size_t)F_ * D_ * 2);   // [D][F]
    }
    bf16* genW1T = (bf16*)alloc((size_t)E_ * D_ * F_ * 2);
    bf16* genW2T = (bf16*)alloc((size_t)E_ * F_ * D_ * 2);
    bf16* hln = (bf16*)alloc((size_t)M_ * D_ * 2);
    bf16* qb  = (bf16*)alloc((size_t)M_ * D_ * 2);
    bf16* kb  = (bf16*)alloc((size_t)M_ * D_ * 2);
    bf16* vb  = (bf16*)alloc((size_t)M_ * D_ * 2);
    bf16* vT  = (bf16*)alloc((size_t)M_ * D_ * 2);
    bf16* att = (bf16*)alloc((size_t)B_ * H_ * S_ * S_ * 2);    // 128 MiB
    bf16* aob = (bf16*)alloc((size_t)M_ * D_ * 2);
    float* x1  = (float*)alloc((size_t)M_ * D_ * 4);
    float* xn  = (float*)alloc((size_t)M_ * D_ * 4);
    bf16* eln  = (bf16*)alloc((size_t)M_ * D_ * 2);
    bf16* h1b  = (bf16*)alloc((size_t)M_ * F_ * 2);             // 32 MiB
    float* comb = (float*)alloc((size_t)M_ * E_ * 4);
    float* vm = (float*)alloc((size_t)M_ * 4);
    float* tm = (float*)alloc((size_t)M_ * 4);

    // ---- output layout: [out | logits | probs | att_mean] ----
    float* outF    = (float*)d_out;
    float* logits  = outF + M_ * D_;
    float* probs   = logits + M_ * E_;
    float* attMean = probs + M_ * E_;

    const dim3 blk(256, 1, 1);
    auto grN = [](long long n) { long long g = (n + 255) / 256; return (unsigned)(g > 8192 ? 8192 : g); };

    // 1) weight downconvert + transpose to bf16 [N][K]
    k_wT<<<grN((long long)D_ * D_), blk, 0, stream>>>(Wq, WqT, D_, D_);
    k_wT<<<grN((long long)D_ * D_), blk, 0, stream>>>(Wk, WkT, D_, D_);
    k_wT<<<grN((long long)D_ * D_), blk, 0, stream>>>(Wv, WvT, D_, D_);
    k_wT<<<grN((long long)D_ * D_), blk, 0, stream>>>(Wo, WoT, D_, D_);
    for (int p = 0; p < 3; ++p) {
        const float* W1 = (const float*)d_in[24 + p * 6];
        const float* W2 = (const float*)d_in[26 + p * 6];
        k_wT<<<grN((long long)D_ * F_), blk, 0, stream>>>(W1, expW1T[p], D_, F_);
        k_wT<<<grN((long long)F_ * D_), blk, 0, stream>>>(W2, expW2T[p], F_, D_);
    }
    for (int e = 0; e < E_; ++e) {
        k_wT<<<grN((long long)D_ * F_), blk, 0, stream>>>(gen_W1 + (long long)e * D_ * F_, genW1T + (long long)e * F_ * D_, D_, F_);
        k_wT<<<grN((long long)F_ * D_), blk, 0, stream>>>(gen_W2 + (long long)e * F_ * D_, genW2T + (long long)e * D_ * F_, F_, D_);
    }

    // 2) LN1 -> bf16
    k_ln<<<dim3((unsigned)M_), blk, 0, stream>>>(x, n1g, n1b, hln, (float*)nullptr, D_);

    // 3) QKV projections (bf16 out, bias); N=1024 -> NJ=4 (block tile 128x128)
    {
        const dim3 g(D_ / 128, (unsigned)(M_ / 128), 1);
        k_gemm<4, false, false, true, false><<<g, blk, 0, stream>>>(hln, WqT, bq, nullptr, qb,
            (int)M_, D_, D_, D_, D_, D_, 1.0f, nullptr, 0, 1, 0, 0, 0, 0, 0, 0);
        k_gemm<4, false, false, true, false><<<g, blk, 0, stream>>>(hln, WkT, bk, nullptr, kb,
            (int)M_, D_, D_, D_, D_, D_, 1.0f, nullptr, 0, 1, 0, 0, 0, 0, 0, 0);
        k_gemm<4, false, false, true, false><<<g, blk, 0, stream>>>(hln, WvT, bv, nullptr, vb,
            (int)M_, D_, D_, D_, D_, D_, 1.0f, nullptr, 0, 1, 0, 0, 0, 0, 0, 0);
    }
    k_vtrans<<<grN(M_ * D_), blk, 0, stream>>>(vb, vT, B_, S_, H_, HD_);

    // 4) scores = q @ k^T / sqrt(hd), batched over (b,h); BT(=k rows) layout is natural
    {
        const dim3 g(S_ / 128, S_ / 128, B_ * H_);
        k_gemm<4, false, false, true, false><<<g, blk, 0, stream>>>(qb, kb, nullptr, nullptr, att,
            S_, S_, HD_, D_, D_, S_, 0.125f, nullptr, 0, H_,
            (long long)S_ * D_, HD_, (long long)S_ * D_, HD_,
            (long long)H_ * S_ * S_, (long long)S_ * S_);
    }
    // 5) softmax in place (bf16) + mean over heads
    k_softmax_att<<<dim3((unsigned)(B_ * S_)), blk, 0, stream>>>(att, attMean, S_, H_);

    // 6) ao = att @ v, batched; N=64 -> NJ=2 (block tile 128x64)
    {
        const dim3 g(HD_ / 64, S_ / 128, B_ * H_);
        k_gemm<2, false, false, true, false><<<g, blk, 0, stream>>>(att, vT, nullptr, nullptr, aob,
            S_, HD_, S_, S_, S_, D_, 1.0f, nullptr, 0, H_,
            (long long)H_ * S_ * S_, (long long)S_ * S_,
            (long long)H_ * HD_ * S_, (long long)HD_ * S_,
            (long long)S_ * D_, (long long)HD_);
    }
    // 7) x1 = x + ao @ Wo + bo
    k_copy_f32<<<grN(M_ * D_), blk, 0, stream>>>(x, x1, M_ * D_);
    {
        const dim3 g(D_ / 128, (unsigned)(M_ / 128), 1);
        k_gemm<4, false, true, false, true><<<g, blk, 0, stream>>>(aob, WoT, bo, x1, nullptr,
            (int)M_, D_, D_, D_, D_, D_, 1.0f, nullptr, 0, 1, 0, 0, 0, 0, 0, 0);
    }

    // 8) LN2 -> xn (f32)
    k_ln<<<dim3((unsigned)M_), blk, 0, stream>>>(x1, n2g, n2b, (bf16*)nullptr, xn, D_);

    // 9) router: logits, probs, comb; modality masks
    k_router<<<grN(M_ * E_), blk, 0, stream>>>(xn, Wr, br, logits, D_, E_, M_);
    k_topk<<<grN(M_), blk, 0, stream>>>(logits, probs, comb, E_, M_);
    k_modmask<<<grN(M_), blk, 0, stream>>>(mt, vm, tm, M_);

    // 10) out = x1 + sum(expert contributions)  (accumulate in-place in d_out)
    k_copy_f32<<<grN(M_ * D_), blk, 0, stream>>>(x1, outF, M_ * D_);

    struct Exp { const float* lng; const float* lnb; const bf16* W1T; const float* b1;
                 const bf16* W2T; const float* b2; const float* rs; int rsStride; };
    Exp ex[11];
    for (int p = 0; p < 3; ++p) {
        ex[p].lng = (const float*)d_in[22 + p * 6];
        ex[p].lnb = (const float*)d_in[23 + p * 6];
        ex[p].W1T = expW1T[p]; ex[p].b1 = (const float*)d_in[25 + p * 6];
        ex[p].W2T = expW2T[p]; ex[p].b2 = (const float*)d_in[27 + p * 6];
        ex[p].rs = (p == 0) ? nullptr : (p == 1 ? vm : tm);
        ex[p].rsStride = 1;
    }
    for (int e = 0; e < E_; ++e) {
        ex[3 + e].lng = gen_lng + (long long)e * D_;
        ex[3 + e].lnb = gen_lnb + (long long)e * D_;
        ex[3 + e].W1T = genW1T + (long long)e * F_ * D_;
        ex[3 + e].b1  = gen_b1 + (long long)e * F_;
        ex[3 + e].W2T = genW2T + (long long)e * D_ * F_;
        ex[3 + e].b2  = gen_b2 + (long long)e * D_;
        ex[3 + e].rs  = comb + e; ex[3 + e].rsStride = E_;
    }

    const dim3 gFC1(F_ / 128, (unsigned)(M_ / 128), 1);
    const dim3 gFC2(D_ / 128, (unsigned)(M_ / 128), 1);
    for (int i = 0; i < 11; ++i) {
        // per-expert LN of xn -> bf16
        k_ln<<<dim3((unsigned)M_), blk, 0, stream>>>(xn, ex[i].lng, ex[i].lnb, eln, (float*)nullptr, D_);
        // fc1 + exact GELU -> bf16
        k_gemm<4, true, false, true, false><<<gFC1, blk, 0, stream>>>(eln, ex[i].W1T, ex[i].b1,
            nullptr, h1b, (int)M_, F_, D_, D_, D_, F_, 1.0f, nullptr, 0, 1, 0, 0, 0, 0, 0, 0);
        // fc2 (+bias), scaled per-row (modality mask / router weight), accumulated into out
        k_gemm<4, false, true, false, true><<<gFC2, blk, 0, stream>>>(h1b, ex[i].W2T, ex[i].b2,
            outF, nullptr, (int)M_, D_, F_, F_, F_, D_, 1.0f, ex[i].rs, ex[i].rsStride,
            1, 0, 0, 0, 0, 0, 0);
    }
}
